// MultiHeadAttention_59949153517921
// MI455X (gfx1250) — matrix operand order repaired and checked by reading, not compiled, not run on hardware
//
#include <hip/hip_runtime.h>

#define D_MODEL 1024
#define NH 16
#define DK 64
#define BATCH 2
#define SEQ 2048
#define MTOT (BATCH*SEQ)   // 4096 rows for all projection GEMMs

typedef __attribute__((ext_vector_type(16))) _Float16 v16h;
typedef __attribute__((ext_vector_type(8)))  _Float16 v8h;
typedef __attribute__((ext_vector_type(4)))  _Float16 v4h;
typedef __attribute__((ext_vector_type(8)))  float    v8f;
typedef __attribute__((ext_vector_type(4)))  float    v4f;

static __device__ __forceinline__ v16h cat8(v8h lo, v8h hi) {
  return __builtin_shufflevector(lo, hi, 0,1,2,3,4,5,6,7,8,9,10,11,12,13,14,15);
}

// A fragment (16 rows x 32 k, f16) from row-major storage.
// Lane l holds row l%16; element e -> k = grp*16 + e.
static __device__ __forceinline__ v16h load_a(const _Float16* __restrict__ base,
                                              int stride, int row, int grp) {
  const _Float16* p = base + (size_t)row * stride + grp * 16;
  return cat8(*(const v8h*)p, *(const v8h*)(p + 8));
}

// B fragment (32 k x 16 cols, f16) from B^T-row-major storage (row = output col n,
// contiguous over k). Lane l holds col l%16; element e -> k = grp*16 + e.
static __device__ __forceinline__ v16h load_b(const _Float16* __restrict__ base,
                                              int stride, int row, int grp) {
  return *(const v16h*)(base + (size_t)row * stride + grp * 16);
}

static __device__ __forceinline__ v8f wmma16(v16h a, v16h b, v8f c) {
  return __builtin_amdgcn_wmma_f32_16x16x32_f16(false, a, false, b, (short)0, c,
                                                false, false);
}

// ---------------------------------------------------------------------------
// f32 -> f16 conversion (vectorized x4)
// ---------------------------------------------------------------------------
__global__ __launch_bounds__(256) void cvt_kernel(const float* __restrict__ s,
                                                  _Float16* __restrict__ d, int n) {
  int i = (blockIdx.x * 256 + threadIdx.x) * 4;
  if (i < n) {
    v4f v = *(const v4f*)(s + i);
    v4h h;
    h[0] = (_Float16)v[0]; h[1] = (_Float16)v[1];
    h[2] = (_Float16)v[2]; h[3] = (_Float16)v[3];
    *(v4h*)(d + i) = h;
  }
}

// mask (int 0/1) -> additive bias (0 / -1e30)
__global__ __launch_bounds__(256) void mask_kernel(const int* __restrict__ m,
                                                   float* __restrict__ mb, int n) {
  int i = blockIdx.x * 256 + threadIdx.x;
  if (i < n) mb[i] = m[i] ? 0.0f : -1e30f;
}

// ---------------------------------------------------------------------------
// GEMM: out = A[4096,1024] @ W^T[1024,1024] + bias.  W is [out,in] row-major,
// which is exactly B^T-row-major for WMMA B fragments.
// mode 0: f32 store (final projection) -> out32
// mode 1: f16 store head-major  [B,H,L,dk]   (Q, K)
// mode 2: f16 store head-transposed [B,H,dk,L] (V)
// Per wave: 64x64 tile, 16 accumulators, double-buffered fragments so 16 b128
// loads are in flight behind every 16-WMMA burst. Block = 8 waves -> 128x256.
// ---------------------------------------------------------------------------
__global__ __launch_bounds__(256) void gemm_kernel(
    const _Float16* __restrict__ A, const _Float16* __restrict__ W,
    const float* __restrict__ bias, float* __restrict__ out32,
    _Float16* __restrict__ out16, int mode) {
  const int tid  = threadIdx.x;
  const int wave = tid >> 5, lane = tid & 31, row = lane & 15, grp = lane >> 4;
  const int m0 = blockIdx.y * 128 + (wave >> 2) * 64;
  const int n0 = blockIdx.x * 256 + (wave & 3) * 64;

  const v8f vz = {0.f, 0.f, 0.f, 0.f, 0.f, 0.f, 0.f, 0.f};
  v8f acc[4][4];
#pragma unroll
  for (int i = 0; i < 4; i++)
#pragma unroll
    for (int j = 0; j < 4; j++) acc[i][j] = vz;

  auto load_frags = [&](int k0, v16h* af, v16h* bf) {
#pragma unroll
    for (int j = 0; j < 4; j++)
      bf[j] = load_b(W + (size_t)(n0 + 16 * j) * D_MODEL + k0, D_MODEL, row, grp);
#pragma unroll
    for (int i = 0; i < 4; i++)
      af[i] = load_a(A + (size_t)(m0 + 16 * i) * D_MODEL + k0, D_MODEL, row, grp);
  };
  auto mma44 = [&](const v16h* af, const v16h* bf) {
#pragma unroll
    for (int i = 0; i < 4; i++)
#pragma unroll
      for (int j = 0; j < 4; j++) acc[i][j] = wmma16(af[i], bf[j], acc[i][j]);
  };

  v16h afA[4], bfA[4], afB[4], bfB[4];
  load_frags(0, afA, bfA);
  // steady state: loads for the next 32-k slice issue before this slice's WMMAs
  for (int k0 = 0; k0 < D_MODEL - 64; k0 += 64) {
    load_frags(k0 + 32, afB, bfB);
    mma44(afA, bfA);
    load_frags(k0 + 64, afA, bfA);
    mma44(afB, bfB);
  }
  load_frags(D_MODEL - 32, afB, bfB);
  mma44(afA, bfA);
  mma44(afB, bfB);

  // --- epilogue: strength-reduced stores -----------------------------------
  float bn[4];
#pragma unroll
  for (int j = 0; j < 4; j++) bn[j] = bias[n0 + 16 * j + row];

  // C layout: VGPR r, lanes 0-15 -> (M = r, N = lane); lanes 16-31 -> (M = 8+r, N = lane-16)
  const int bb = m0 >> 11;          // batch (tile never crosses batch boundary)
  const int l0 = m0 & (SEQ - 1);
  const int h  = n0 >> 6;           // wave tile spans exactly one head (dk = 64)
  if (mode == 0) {
    float* bp = out32 + (size_t)(m0 + 8 * grp) * D_MODEL + n0 + row;
#pragma unroll
    for (int i = 0; i < 4; i++)
#pragma unroll
      for (int j = 0; j < 4; j++)
#pragma unroll
        for (int r = 0; r < 8; r++)
          bp[(size_t)(16 * i + r) * D_MODEL + 16 * j] = acc[i][j][r] + bn[j];
  } else if (mode == 1) {
    _Float16* bp = out16 + (((size_t)(bb * NH + h)) * SEQ + l0 + 8 * grp) * DK + row;
#pragma unroll
    for (int i = 0; i < 4; i++)
#pragma unroll
      for (int j = 0; j < 4; j++)
#pragma unroll
        for (int r = 0; r < 8; r++)
          bp[(size_t)(16 * i + r) * DK + 16 * j] = (_Float16)(acc[i][j][r] + bn[j]);
  } else {
    _Float16* bp = out16 + (((size_t)(bb * NH + h)) * DK + row) * SEQ + l0 + 8 * grp;
#pragma unroll
    for (int i = 0; i < 4; i++)
#pragma unroll
      for (int j = 0; j < 4; j++)
#pragma unroll
        for (int r = 0; r < 8; r++)
          bp[(size_t)(16 * j) * SEQ + 16 * i + r] = (_Float16)(acc[i][j][r] + bn[j]);
  }
}

// ---------------------------------------------------------------------------
// Flash attention, one (b,h) per blockIdx.y, 128 queries per block (16 / wave).
// Computes S^T = K Q^T so that softmaxed P^T converts to a WMMA B fragment
// with only per-lane packs + shfl_xor(16); then O^T = V^T P^T accumulates.
// K/V fragments are double-buffered one 32-key block ahead so global loads
// overlap the exp2/pack VALU chain.
// ---------------------------------------------------------------------------
__global__ __launch_bounds__(256) void attn_kernel(
    const _Float16* __restrict__ Q, const _Float16* __restrict__ K,
    const _Float16* __restrict__ Vt, const float* __restrict__ mbias,
    _Float16* __restrict__ comb) {
  const int tid  = threadIdx.x;
  const int wave = tid >> 5, lane = tid & 31, row = lane & 15, grp = lane >> 4;
  const int bh = blockIdx.y, bb = bh >> 4, h = bh & 15;
  const int m0 = blockIdx.x * 128 + wave * 16;

  const _Float16* Qh = Q + (size_t)bh * SEQ * DK;
  const _Float16* Kh = K + (size_t)bh * SEQ * DK;
  const _Float16* Vh = Vt + (size_t)bh * DK * SEQ;
  const float* mb = mbias + (size_t)bb * SEQ;

  // Q^T B-fragments (col = query m, k = d), loaded once.
  const v16h qb0 = load_b(Qh + (size_t)m0 * DK + 0,  DK, row, grp);
  const v16h qb1 = load_b(Qh + (size_t)m0 * DK + 32, DK, row, grp);

  const v8f vz = {0.f, 0.f, 0.f, 0.f, 0.f, 0.f, 0.f, 0.f};
  v8f o[4];  // O^T accumulators, d-subtiles 0,16,32,48; cols = 16 queries
  o[0] = vz; o[1] = vz; o[2] = vz; o[3] = vz;
  float mrun = -1e30f, lrun = 0.0f;
  const float cf = 0.18033688011112042f;  // log2(e)/sqrt(dk), exp2-domain softmax

  auto loadKV = [&](int n0, v16h* ka, v16h* va) {
    ka[0] = load_a(Kh + (size_t)n0 * DK + 0,         DK, row, grp);
    ka[1] = load_a(Kh + (size_t)n0 * DK + 32,        DK, row, grp);
    ka[2] = load_a(Kh + (size_t)(n0 + 16) * DK + 0,  DK, row, grp);
    ka[3] = load_a(Kh + (size_t)(n0 + 16) * DK + 32, DK, row, grp);
#pragma unroll
    for (int t = 0; t < 4; t++)
      va[t] = load_a(Vh + (size_t)(t * 16) * SEQ + n0, SEQ, row, grp);
  };

  auto process = [&](int n0, const v16h* ka, const v16h* va) {
    // S^T tiles: s0 -> keys n0..n0+15, s1 -> keys n0+16..n0+31 (16 queries wide)
    v8f s0 = vz, s1 = vz;
    s0 = wmma16(ka[0], qb0, s0);
    s0 = wmma16(ka[1], qb1, s0);
    s1 = wmma16(ka[2], qb0, s1);
    s1 = wmma16(ka[3], qb1, s1);
    // mask bias per element: s0[r] -> key n0 + r + 8*grp ; s1[r] -> +16
    v4f mb0a = *(const v4f*)(mb + n0 + grp * 8);
    v4f mb0b = *(const v4f*)(mb + n0 + grp * 8 + 4);
    v4f mb1a = *(const v4f*)(mb + n0 + 16 + grp * 8);
    v4f mb1b = *(const v4f*)(mb + n0 + 16 + grp * 8 + 4);
    float sv0[8], sv1[8];
#pragma unroll
    for (int r = 0; r < 8; r++) {
      sv0[r] = s0[r] * cf + (r < 4 ? mb0a[r] : mb0b[r - 4]);
      sv1[r] = s1[r] * cf + (r < 4 ? mb1a[r] : mb1b[r - 4]);
    }
    // online softmax: row (query) data split across lane pair (l, l+16)
    float bm = sv0[0];
#pragma unroll
    for (int r = 1; r < 8; r++) bm = fmaxf(bm, sv0[r]);
#pragma unroll
    for (int r = 0; r < 8; r++) bm = fmaxf(bm, sv1[r]);
    bm = fmaxf(bm, __shfl_xor(bm, 16, 32));
    const float mnew = fmaxf(mrun, bm);
    const float corr = exp2f(mrun - mnew);
    float p0[8], p1[8], bsum = 0.f;
#pragma unroll
    for (int r = 0; r < 8; r++) {
      p0[r] = exp2f(sv0[r] - mnew);
      p1[r] = exp2f(sv1[r] - mnew);
      bsum += p0[r] + p1[r];
    }
    bsum += __shfl_xor(bsum, 16, 32);
    lrun = lrun * corr + bsum;
    mrun = mnew;
#pragma unroll
    for (int t = 0; t < 4; t++)
#pragma unroll
      for (int r = 0; r < 8; r++) o[t][r] *= corr;

    // P^T -> B fragment: pack to f16 pairs, exchange with partner lane (xor 16).
    unsigned own0[4], own1[4], par0[4], par1[4];
#pragma unroll
    for (int i = 0; i < 4; i++) {
      own0[i] = __builtin_bit_cast(unsigned, __builtin_amdgcn_cvt_pkrtz(p0[2 * i], p0[2 * i + 1]));
      own1[i] = __builtin_bit_cast(unsigned, __builtin_amdgcn_cvt_pkrtz(p1[2 * i], p1[2 * i + 1]));
    }
#pragma unroll
    for (int i = 0; i < 4; i++) {
      par0[i] = __shfl_xor(own0[i], 16, 32);
      par1[i] = __shfl_xor(own1[i], 16, 32);
    }
    union { v16h hv; unsigned u[8]; } pf;
#pragma unroll
    for (int i = 0; i < 4; i++) {
      pf.u[i]     = grp ? par1[i] : own0[i];  // k = n0 + grp*16 + 0..7
      pf.u[4 + i] = grp ? own1[i] : par0[i];  // k = n0 + grp*16 + 8..15
    }
    // O^T += V^T(tile) x P^T ; V^T stored [dk, L] so A fragments load contiguously.
#pragma unroll
    for (int t = 0; t < 4; t++) o[t] = wmma16(va[t], pf.hv, o[t]);
  };

  v16h kaA[4], vaA[4], kaB[4], vaB[4];
  loadKV(0, kaA, vaA);
  for (int n0 = 0; n0 < SEQ - 64; n0 += 64) {
    loadKV(n0 + 32, kaB, vaB);
    process(n0, kaA, vaA);
    loadKV(n0 + 64, kaA, vaA);
    process(n0 + 32, kaB, vaB);
  }
  loadKV(SEQ - 32, kaB, vaB);
  process(SEQ - 64, kaA, vaA);
  process(SEQ - 32, kaB, vaB);

  const float inv = 1.0f / lrun;
  // O^T C-layout: lane -> query m = m0 + row; VGPR r -> d = t*16 + r + 8*grp
  _Float16* cp = comb + ((size_t)bb * SEQ + m0 + row) * D_MODEL + h * DK;
#pragma unroll
  for (int t = 0; t < 4; t++)
#pragma unroll
    for (int r = 0; r < 8; r++)
      cp[t * 16 + r + 8 * grp] = (_Float16)(o[t][r] * inv);
}

// ---------------------------------------------------------------------------
extern "C" void kernel_launch(void* const* d_in, const int* in_sizes, int n_in,
                              void* d_out, int out_size, void* d_ws, size_t ws_size,
                              hipStream_t stream) {
  (void)in_sizes; (void)n_in; (void)out_size; (void)ws_size;
  const float* query = (const float*)d_in[0];
  const float* key   = (const float*)d_in[1];
  const float* value = (const float*)d_in[2];
  const int*   mask  = (const int*)d_in[3];
  const float* Wq = (const float*)d_in[4];
  const float* bq = (const float*)d_in[5];
  const float* Wk = (const float*)d_in[6];
  const float* bk = (const float*)d_in[7];
  const float* Wv = (const float*)d_in[8];
  const float* bv = (const float*)d_in[9];
  const float* Wo = (const float*)d_in[10];
  const float* bo = (const float*)d_in[11];

  char* ws = (char*)d_ws;
  const size_t SZ_X = (size_t)MTOT * D_MODEL * sizeof(_Float16);      // 8 MiB
  const size_t SZ_W = (size_t)D_MODEL * D_MODEL * sizeof(_Float16);   // 2 MiB
  _Float16* x16q = (_Float16*)(ws);
  _Float16* x16k = (_Float16*)(ws + SZ_X);
  _Float16* x16v = (_Float16*)(ws + 2 * SZ_X);
  _Float16* wq16 = (_Float16*)(ws + 3 * SZ_X);
  _Float16* wk16 = (_Float16*)(ws + 3 * SZ_X + SZ_W);
  _Float16* wv16 = (_Float16*)(ws + 3 * SZ_X + 2 * SZ_W);
  _Float16* wo16 = (_Float16*)(ws + 3 * SZ_X + 3 * SZ_W);
  _Float16* Qh   = (_Float16*)(ws + 3 * SZ_X + 4 * SZ_W);
  _Float16* Kh   = (_Float16*)(ws + 4 * SZ_X + 4 * SZ_W);
  _Float16* Vt   = (_Float16*)(ws + 5 * SZ_X + 4 * SZ_W);
  float*    mbias = (float*)(ws + 6 * SZ_X + 4 * SZ_W);
  _Float16* comb = x16q;  // x16q is dead after the Q projection -> reuse for attn output

  const int nX = MTOT * D_MODEL;      // 4,194,304 (multiple of 1024)
  const int nW = D_MODEL * D_MODEL;   // 1,048,576
  cvt_kernel<<<nX / 1024, 256, 0, stream>>>(query, x16q, nX);
  cvt_kernel<<<nX / 1024, 256, 0, stream>>>(key,   x16k, nX);
  cvt_kernel<<<nX / 1024, 256, 0, stream>>>(value, x16v, nX);
  cvt_kernel<<<nW / 1024, 256, 0, stream>>>(Wq, wq16, nW);
  cvt_kernel<<<nW / 1024, 256, 0, stream>>>(Wk, wk16, nW);
  cvt_kernel<<<nW / 1024, 256, 0, stream>>>(Wv, wv16, nW);
  cvt_kernel<<<nW / 1024, 256, 0, stream>>>(Wo, wo16, nW);
  mask_kernel<<<MTOT / 256, 256, 0, stream>>>(mask, mbias, MTOT);

  dim3 gg(D_MODEL / 256, MTOT / 128);  // (4, 32)
  gemm_kernel<<<gg, 256, 0, stream>>>(x16q, wq16, bq, nullptr, Qh, 1);
  gemm_kernel<<<gg, 256, 0, stream>>>(x16k, wk16, bk, nullptr, Kh, 1);
  gemm_kernel<<<gg, 256, 0, stream>>>(x16v, wv16, bv, nullptr, Vt, 2);

  dim3 ga(SEQ / 128, BATCH * NH);      // (16, 32)
  attn_kernel<<<ga, 256, 0, stream>>>(Qh, Kh, Vt, mbias, comb);

  gemm_kernel<<<gg, 256, 0, stream>>>(comb, wo16, bo, (float*)d_out, nullptr, 0);
}
